// AttentionLoRA_12584254177757
// MI455X (gfx1250) — matrix operand order repaired and checked by reading, not compiled, not run on hardware
//
#include <hip/hip_runtime.h>

// ---------------------------------------------------------------------------
// Fused causal attention + LoRA (merged weights) for MI455X / gfx1250.
//  * all intermediates f16; WMMA = v_wmma_f32_16x16x32_f16 (f32 accum)
//  * GEMM + attention K-staging use the Tensor Data Mover
//    (tensor_load_to_lds + s_wait_tensorcnt), double-buffered in the GEMM.
// ---------------------------------------------------------------------------

typedef __attribute__((ext_vector_type(16))) _Float16 v16h;
typedef __attribute__((ext_vector_type(8)))  _Float16 v8h;
typedef __attribute__((ext_vector_type(4)))  _Float16 v4h;
typedef __attribute__((ext_vector_type(8)))  float    v8f;
typedef __attribute__((ext_vector_type(4)))  unsigned int u32x4;
typedef __attribute__((ext_vector_type(8)))  int      i32x8;
typedef __attribute__((ext_vector_type(4)))  int      i32x4;

#define WMMA_F16(a, b, c) \
  __builtin_amdgcn_wmma_f32_16x16x32_f16(false, (a), false, (b), (short)0, (c), false, false)

static constexpr int B_ = 2, T_ = 2048, C_ = 1024, H_ = 16, D_ = 64, R_ = 16;
static constexpr int M_ = B_ * T_;   // 4096 token rows

// ---------------------------------------------------------------------------
// TDM helper: 2D tile load (f16 elements) global -> LDS with row padding.
// Descriptor layout per CDNA5 ISA ch.8 (group0: count/lds/global/type,
// group1: data_size, pad, tensor dims, tile dims, dim0 stride).
// ---------------------------------------------------------------------------
__device__ __forceinline__ unsigned lds_off(const void* p) {
  // generic pointers to LDS carry the byte offset in the low 32 bits
  return (unsigned)(uintptr_t)p;
}

__device__ __forceinline__ void tdm_load_2d_f16(
    unsigned lds_byte_off, const _Float16* gptr,
    int tile_d0,            // elements along dim0 (contiguous)
    int tile_d1,            // rows
    long long stride_elems, // global row stride in elements
    int pad_interval_code,  // DWORDs-per-pad code: 0=2,1=4,2=8,3=16,4=32,...
    int pad_amount_code) {  // pad DWORDs-1
  unsigned long long ga = (unsigned long long)(uintptr_t)gptr;
  u32x4 g0;
  g0[0] = 1u;                                            // count=1 (valid D#)
  g0[1] = lds_byte_off;                                  // lds_addr
  g0[2] = (unsigned)(ga & 0xFFFFFFFFull);                // global_addr[31:0]
  g0[3] = (unsigned)((ga >> 32) & 0x1FFFFFFull) | (2u << 30);  // addr[56:32]|type=2
  i32x8 g1;
  unsigned w0 = (1u << 16)                               // data_size = 2 bytes
              | (1u << 20)                               // pad_enable
              | ((unsigned)pad_interval_code << 22)
              | ((unsigned)pad_amount_code << 25);
  g1[0] = (int)w0;
  g1[1] = (int)(((unsigned)tile_d0 & 0xFFFFu) << 16);    // tensor_dim0[15:0]
  g1[2] = (int)((((unsigned)tile_d0 >> 16) & 0xFFFFu) |
                (((unsigned)tile_d1 & 0xFFFFu) << 16));  // dim0[31:16]|dim1[15:0]
  g1[3] = (int)((((unsigned)tile_d1 >> 16) & 0xFFFFu) |
                (((unsigned)tile_d0 & 0xFFFFu) << 16));  // dim1[31:16]|tile_dim0
  g1[4] = (int)((unsigned)tile_d1 & 0xFFFFu);            // tile_dim1 (tile_dim2=0)
  g1[5] = (int)(unsigned)(stride_elems & 0xFFFFFFFFll);  // dim0_stride[31:0]
  g1[6] = (int)(unsigned)((stride_elems >> 32) & 0xFFFFll); // dim0_stride[47:32]
  g1[7] = 0;
  i32x4 z4 = {};
#if __has_include(<hip/amd_detail/amd_gfx1250_TDM.h>)
  i32x8 z8 = {};
  __builtin_amdgcn_tensor_load_to_lds(g0, g1, z4, z4, z8, 0);   // clang-23 form
#else
  __builtin_amdgcn_tensor_load_to_lds(g0, g1, z4, z4, 0);       // ROCm 7.2 form
#endif
}

// ---------------------------------------------------------------------------
// f32 -> f16 conversion (n multiple of 4)
// ---------------------------------------------------------------------------
__global__ void __launch_bounds__(256)
cvt_f32_to_f16(const float* __restrict__ in, _Float16* __restrict__ out, int n) {
  int i = blockIdx.x * blockDim.x + threadIdx.x;
  if (i * 4 >= n) return;
  float4 v = ((const float4*)in)[i];
  v4h o; o[0] = (_Float16)v.x; o[1] = (_Float16)v.y;
  o[2] = (_Float16)v.z; o[3] = (_Float16)v.w;
  *(v4h*)(out + i * 4) = o;
}

// ---------------------------------------------------------------------------
// LoRA merge to f16: Wm[n,k] = f16(W[n,k] + scale * sum_r Blo[n,r]*Alo[r,k])
// ---------------------------------------------------------------------------
__global__ void __launch_bounds__(256)
lora_merge_f16(const float* __restrict__ W, const float* __restrict__ Blo,
               const float* __restrict__ Alo, _Float16* __restrict__ Wm,
               int N, int K, float scale) {
  int idx = blockIdx.x * blockDim.x + threadIdx.x;
  if (idx >= N * K) return;
  int n = idx / K, k = idx - n * K;
  float acc = 0.f;
#pragma unroll
  for (int r = 0; r < R_; ++r) acc += Blo[n * R_ + r] * Alo[r * K + k];
  Wm[idx] = (_Float16)(W[idx] + scale * acc);
}

// ---------------------------------------------------------------------------
// GEMM: Cout[M,N] = A[M,K] @ W[N,K]^T + bias[N]
//  A, W f16; out f16 or f32 (template). Block = 256 threads (8 waves),
//  block tile 128x128, K-step 32, TDM double-buffered tile staging.
//  Wave (wm,wn) computes 32 rows x 64 cols: 2 A-frags x 4 B-frags, 8 WMMAs.
// ---------------------------------------------------------------------------
#define GBM 128
#define GBN 128
#define GBK 32
#define SA_STR 40   // halves: 32 data + 8 pad (pad_interval=16dw, pad=4dw)

template <bool OUT_F16>
__global__ void __launch_bounds__(256)
gemm_f16wmma(const _Float16* __restrict__ A, const _Float16* __restrict__ W,
             const float* __restrict__ bias, void* __restrict__ Cout,
             int M, int N, int K) {
  __shared__ __align__(16) _Float16 sA[2][GBM * SA_STR];
  __shared__ __align__(16) _Float16 sW[2][GBN * SA_STR];

  const int tid  = threadIdx.x;
  const int lane = tid & 31;
  const int wave = tid >> 5;
  const int hi   = lane >> 4;
  const int lo   = lane & 15;
  const int wm   = wave >> 1;          // 0..3 -> 32-row band
  const int wn   = wave & 1;           // 0..1 -> 64-col band
  const int m0   = blockIdx.y * GBM;
  const int n0   = blockIdx.x * GBN;
  const int nK   = K / GBK;

  v8f acc[2][4] = {};

  // prologue: stage k-step 0 into buffer 0
  if (wave == 0) {
    tdm_load_2d_f16(lds_off(&sA[0][0]), A + (size_t)m0 * K, GBK, GBM, K, 3, 3);
    tdm_load_2d_f16(lds_off(&sW[0][0]), W + (size_t)n0 * K, GBK, GBN, K, 3, 3);
  }

  for (int kk = 0; kk < nK; ++kk) {
    const int cur = kk & 1;
    if (wave == 0) {
      if (kk + 1 < nK) {
        const int k0n = (kk + 1) * GBK;
        tdm_load_2d_f16(lds_off(&sA[cur ^ 1][0]), A + (size_t)m0 * K + k0n,
                        GBK, GBM, K, 3, 3);
        tdm_load_2d_f16(lds_off(&sW[cur ^ 1][0]), W + (size_t)n0 * K + k0n,
                        GBK, GBN, K, 3, 3);
        __builtin_amdgcn_s_wait_tensorcnt(2);  // current buffer's 2 ops done
      } else {
        __builtin_amdgcn_s_wait_tensorcnt(0);
      }
    }
    __syncthreads();

    // ---- fragment loads (all up-front, then 8 back-to-back WMMAs)
    v16h af[2];
#pragma unroll
    for (int r = 0; r < 2; ++r) {
      const _Float16* base = &sA[cur][(wm * 32 + r * 16 + lo) * SA_STR + hi * 8];
      v4h a0 = *(const v4h*)(base + 0);
      v4h a1 = *(const v4h*)(base + 4);
      v4h a2 = *(const v4h*)(base + 16);
      v4h a3 = *(const v4h*)(base + 20);
#pragma unroll
      for (int i = 0; i < 4; ++i) {
        af[r][i] = a0[i]; af[r][4 + i] = a1[i];
        af[r][8 + i] = a2[i]; af[r][12 + i] = a3[i];
      }
    }
    v16h bf[4];
#pragma unroll
    for (int j = 0; j < 4; ++j) {
      const _Float16* base = &sW[cur][(wn * 64 + j * 16 + lo) * SA_STR + hi * 8];
      v8h b0 = *(const v8h*)(base);
      v8h b1 = *(const v8h*)(base + 16);
#pragma unroll
      for (int i = 0; i < 8; ++i) { bf[j][i] = b0[i]; bf[j][8 + i] = b1[i]; }
    }
#pragma unroll
    for (int r = 0; r < 2; ++r)
#pragma unroll
      for (int j = 0; j < 4; ++j)
        acc[r][j] = WMMA_F16(af[r], bf[j], acc[r][j]);
    __syncthreads();
  }

  // ---- writeback: C/D layout row = f + 8*hi, col = lo
#pragma unroll
  for (int r = 0; r < 2; ++r)
#pragma unroll
    for (int j = 0; j < 4; ++j)
#pragma unroll
      for (int f = 0; f < 8; ++f) {
        int row = m0 + wm * 32 + r * 16 + hi * 8 + f;
        int col = n0 + wn * 64 + j * 16 + lo;
        float v = acc[r][j][f] + bias[col];
        if (OUT_F16) ((_Float16*)Cout)[(size_t)row * N + col] = (_Float16)v;
        else         ((float*)Cout)[(size_t)row * N + col] = v;
      }
}

// ---------------------------------------------------------------------------
// Flash attention over f16 qkv [B, T, 3C] -> f16 y [B, T, C].
// Grid (T/64, H, B), 128 threads (4 waves), 32-key chunks.
// K chunk staged by TDM; V staged manually (transposed in LDS).
// ---------------------------------------------------------------------------
#define SK_STR  72   // halves: 64 data + 8 pad (pad_interval=32dw, pad=4dw)
#define SVT_STR 40
#define SP_STR  36
#define SS_STR  33

__global__ void __launch_bounds__(128)
attn_flash_f16wmma(const _Float16* __restrict__ qkv, _Float16* __restrict__ y) {
  __shared__ __align__(16) _Float16 sK[32 * SK_STR];      // [key][d]
  __shared__ __align__(16) _Float16 sVt[64 * SVT_STR];    // [d][key]
  __shared__ __align__(16) _Float16 sP[4 * 16 * SP_STR];  // per-wave P tile
  __shared__ float sS[4 * 16 * SS_STR];                   // per-wave S tile
  __shared__ float sRow[4 * 16];                          // alpha / 1/l bcast

  const int tid  = threadIdx.x;
  const int lane = tid & 31;
  const int wave = tid >> 5;
  const int hi   = lane >> 4;
  const int lo   = lane & 15;
  const int b    = blockIdx.z;
  const int h    = blockIdx.y;
  const int q0   = blockIdx.x * 64;
  const size_t rs = (size_t)3 * C_;
  const float scale = 0.125f;  // 1/sqrt(64)

  // ---- Q fragments in registers for whole kernel
  v16h qf[2];
  {
    const int qrow = q0 + wave * 16 + lo;
    const _Float16* qb = qkv + (size_t)(b * T_ + qrow) * rs + h * D_;
#pragma unroll
    for (int c = 0; c < 2; ++c)
#pragma unroll
      for (int e = 0; e < 16; ++e) {
        int k = (e & 7) + ((e >> 3) << 4) + (hi << 3);
        qf[c][e] = qb[c * 32 + k];
      }
  }

  v8f o[4] = {};
  float m_r = -1e30f, l_r = 0.f;      // lanes 0..15: row = lane

  const int nchunks = (q0 >> 5) + 2;  // keys 0 .. q0+63
  for (int kc = 0; kc < nchunks; ++kc) {
    const int kb = kc * 32;

    // ---- K chunk 32x64 via TDM (wave 0 issues, everyone syncs after)
    if (wave == 0)
      tdm_load_2d_f16(lds_off(sK),
                      qkv + (size_t)(b * T_ + kb) * rs + C_ + h * D_,
                      D_, 32, (long long)rs, 4, 3);
    // ---- V chunk staged transposed: sVt[d][key]
    {
      int key = tid >> 2;
      int db  = (tid & 3) * 16;
      const _Float16* vp = qkv + (size_t)(b * T_ + kb + key) * rs + 2 * C_ + h * D_ + db;
      v8h v0 = *(const v8h*)(vp);
      v8h v1 = *(const v8h*)(vp + 8);
#pragma unroll
      for (int i = 0; i < 8; ++i) {
        sVt[(db + i) * SVT_STR + key]     = v0[i];
        sVt[(db + 8 + i) * SVT_STR + key] = v1[i];
      }
    }
    if (wave == 0) __builtin_amdgcn_s_wait_tensorcnt(0);
    __syncthreads();

    // ---- S = Q @ K^T (two key subtiles, 2 K-dim chunks of 32)
    v8f s0 = {}, s1 = {};
#pragma unroll
    for (int c = 0; c < 2; ++c) {
      const _Float16* k0p = sK + lo * SK_STR + c * 32 + hi * 8;
      const _Float16* k1p = sK + (16 + lo) * SK_STR + c * 32 + hi * 8;
      v8h t0a = *(const v8h*)(k0p), t0b = *(const v8h*)(k0p + 16);
      v8h t1a = *(const v8h*)(k1p), t1b = *(const v8h*)(k1p + 16);
      v16h bf0, bf1;
#pragma unroll
      for (int i = 0; i < 8; ++i) {
        bf0[i] = t0a[i]; bf0[8 + i] = t0b[i];
        bf1[i] = t1a[i]; bf1[8 + i] = t1b[i];
      }
      s0 = WMMA_F16(qf[c], bf0, s0);
      s1 = WMMA_F16(qf[c], bf1, s1);
    }

    // ---- scale + causal mask -> per-wave LDS S tile
    {
      float* srow = sS + wave * 16 * SS_STR;
#pragma unroll
      for (int f = 0; f < 8; ++f) {
        int r = hi * 8 + f;
        int q = q0 + wave * 16 + r;
        float x0 = s0[f] * scale, x1 = s1[f] * scale;
        if (kb + lo > q)      x0 = -1e30f;
        if (kb + 16 + lo > q) x1 = -1e30f;
        srow[r * SS_STR + lo]      = x0;
        srow[r * SS_STR + 16 + lo] = x1;
      }
    }

    // ---- online softmax: lane r<16 owns row r (same-wave LDS in-order)
    if (lane < 16) {
      const float* srow = sS + (wave * 16 + lane) * SS_STR;
      float mx = m_r;
#pragma unroll
      for (int j = 0; j < 32; ++j) mx = fmaxf(mx, srow[j]);
      float alpha = __expf(m_r - mx);
      float sum = 0.f;
      _Float16* prow = sP + (wave * 16 + lane) * SP_STR;
#pragma unroll
      for (int j = 0; j < 32; ++j) {
        float e = __expf(srow[j] - mx);
        sum += e;
        prow[j] = (_Float16)e;
      }
      l_r = l_r * alpha + sum;
      m_r = mx;
      sRow[wave * 16 + lane] = alpha;
    }

    // ---- O *= alpha ; O += P @ V
    {
#pragma unroll
      for (int f = 0; f < 8; ++f) {
        float alpha = sRow[wave * 16 + hi * 8 + f];
#pragma unroll
        for (int j = 0; j < 4; ++j) o[j][f] *= alpha;
      }
      const _Float16* pb = sP + (wave * 16 + lo) * SP_STR + hi * 8;
      v4h p0 = *(const v4h*)(pb + 0);
      v4h p1 = *(const v4h*)(pb + 4);
      v4h p2 = *(const v4h*)(pb + 16);
      v4h p3 = *(const v4h*)(pb + 20);
      v16h pf;
#pragma unroll
      for (int i = 0; i < 4; ++i) {
        pf[i] = p0[i]; pf[4 + i] = p1[i]; pf[8 + i] = p2[i]; pf[12 + i] = p3[i];
      }
#pragma unroll
      for (int j = 0; j < 4; ++j) {
        const _Float16* vb = sVt + (j * 16 + lo) * SVT_STR + hi * 8;
        v8h va = *(const v8h*)(vb);
        v8h vc = *(const v8h*)(vb + 16);
        v16h vf;
#pragma unroll
        for (int i = 0; i < 8; ++i) { vf[i] = va[i]; vf[8 + i] = vc[i]; }
        o[j] = WMMA_F16(pf, vf, o[j]);
      }
    }
    __syncthreads();
  }

  // ---- normalize and write y (f16)
  if (lane < 16) sRow[wave * 16 + lane] = 1.0f / l_r;
#pragma unroll
  for (int f = 0; f < 8; ++f) {
    int r = hi * 8 + f;
    float linv = sRow[wave * 16 + r];
    int q = q0 + wave * 16 + r;
    _Float16* yb = y + (size_t)(b * T_ + q) * C_ + h * D_;
#pragma unroll
    for (int j = 0; j < 4; ++j) yb[j * 16 + lo] = (_Float16)(o[j][f] * linv);
  }
}

// ---------------------------------------------------------------------------
// Launch: cvt x -> f16, merge LoRA -> f16 weights, QKV GEMM, attention, proj
// ---------------------------------------------------------------------------
extern "C" void kernel_launch(void* const* d_in, const int* in_sizes, int n_in,
                              void* d_out, int out_size, void* d_ws, size_t ws_size,
                              hipStream_t stream) {
  (void)in_sizes; (void)n_in; (void)out_size; (void)ws_size;
  const float* x       = (const float*)d_in[0];
  const float* w_attn  = (const float*)d_in[1];
  const float* b_attn  = (const float*)d_in[2];
  const float* la_attn = (const float*)d_in[3];
  const float* lb_attn = (const float*)d_in[4];
  const float* w_proj  = (const float*)d_in[5];
  const float* b_proj  = (const float*)d_in[6];
  const float* la_proj = (const float*)d_in[7];
  const float* lb_proj = (const float*)d_in[8];
  float* out = (float*)d_out;

  // workspace layout (f16 halves, all regions 16B aligned)
  _Float16* xh      = (_Float16*)d_ws;                    // M*C
  _Float16* wm_attn = xh + (size_t)M_ * C_;               // 3C*C
  _Float16* wm_proj = wm_attn + (size_t)3 * C_ * C_;      // C*C
  _Float16* qkv     = wm_proj + (size_t)C_ * C_;          // M*3C
  _Float16* yattn   = qkv + (size_t)M_ * 3 * C_;          // M*C

  const float lscale = 1.0f / (float)R_;

  cvt_f32_to_f16<<<(M_ * C_ / 4 + 255) / 256, 256, 0, stream>>>(x, xh, M_ * C_);
  lora_merge_f16<<<(3 * C_ * C_ + 255) / 256, 256, 0, stream>>>(
      w_attn, lb_attn, la_attn, wm_attn, 3 * C_, C_, lscale);
  lora_merge_f16<<<(C_ * C_ + 255) / 256, 256, 0, stream>>>(
      w_proj, lb_proj, la_proj, wm_proj, C_, C_, lscale);

  gemm_f16wmma<true><<<dim3(3 * C_ / GBN, M_ / GBM), 256, 0, stream>>>(
      xh, wm_attn, b_attn, qkv, M_, 3 * C_, C_);

  attn_flash_f16wmma<<<dim3(T_ / 64, H_, B_), 128, 0, stream>>>(qkv, yattn);

  gemm_f16wmma<false><<<dim3(C_ / GBN, M_ / GBM), 256, 0, stream>>>(
      yattn, wm_proj, b_proj, out, M_, C_, C_);
}
